// RPN_16913581211971
// MI455X (gfx1250) — matrix operand order repaired and checked by reading, not compiled, not run on hardware
//
#include <hip/hip_runtime.h>
#include <hip/hip_bf16.h>
#include <math.h>

typedef __attribute__((ext_vector_type(16))) _Float16 v16h;
typedef __attribute__((ext_vector_type(8)))  float    v8f;

#define Hc 64
#define Wc 64
#define Cc 512
#define Bc 2
#define NA 36864      // 64*64*9 anchors
#define POST 300

// ---------------------------------------------------------------------------
// Kernel 1: 3x3 SAME conv + bias + ReLU as implicit GEMM on WMMA (f16 in, f32 acc)
// grid = (64 rows, 512/32 co-tiles, 2 batches), block = 256 threads (8 waves)
// Block macro-tile: 32 (Cout) x 64 (one full image row). Each wave owns a
// 16x16 f32 accumulator; K = 512*9 reduced in 144 steps of 32.
// Output stored (B,H,W,C) so the 1x1 heads read contiguous K.
// ---------------------------------------------------------------------------
__global__ __launch_bounds__(256) void conv3x3_wmma(
    const float* __restrict__ feat,   // (B,512,64,64)
    const float* __restrict__ wgt,    // (512,512,3,3) OIHW
    const float* __restrict__ bias,   // (512)
    float* __restrict__ out)          // (B,64,64,512)
{
  __shared__ _Float16 Als[32][32];    // [co_local][k_local]
  __shared__ _Float16 Bls[32][72];    // [k_local][x] (+pad vs bank conflicts)

  const int y    = blockIdx.x;
  const int cob  = blockIdx.y * 32;
  const int b    = blockIdx.z;
  const int t    = threadIdx.x;
  const int wid  = t >> 5;            // wave32
  const int lane = t & 31;
  const int coSub = (wid >> 2) * 16;  // 0 / 16
  const int pxSub = (wid & 3) * 16;   // 0/16/32/48

  v8f acc = {};

  for (int tap = 0; tap < 9; ++tap) {
    const int kh = tap / 3, kw = tap % 3;
    const int gy = y + kh - 1;
    for (int cib = 0; cib < Cc; cib += 32) {
      // ---- stage A tile (weights) 32x32 into LDS, fp32 -> f16
      for (int e = t; e < 32 * 32; e += 256) {
        int co_l = e >> 5, ci_l = e & 31;
        float v = wgt[((size_t)(cob + co_l) * Cc + (cib + ci_l)) * 9 + tap];
        Als[co_l][ci_l] = (_Float16)v;
      }
      // ---- stage B tile (im2col feature row) 32x64 into LDS
      for (int e = t; e < 32 * 64; e += 256) {
        int ci_l = e >> 6, x = e & 63;
        int gx = x + kw - 1;
        float v = 0.f;
        if (gy >= 0 && gy < Hc && gx >= 0 && gx < Wc)
          v = feat[(((size_t)b * Cc + (cib + ci_l)) * Hc + gy) * Wc + gx];
        Bls[ci_l][x] = (_Float16)v;
      }
      // prefetch next Ci block of this row into the cache hierarchy
      if (cib + 32 < Cc && gy >= 0 && gy < Hc)
        __builtin_prefetch(
            feat + (((size_t)b * Cc + (cib + 32 + lane)) * Hc + gy) * Wc, 0, 1);
      __syncthreads();

      // ---- build fragments per documented wave32 VGPR striping
      // A 16x32 f16: lanes 0-15 -> M=lane, K pairs {0..7,16..23};
      //              lanes 16-31 -> M=lane-16, K pairs {8..15,24..31}
      v16h af, bf;
      const int m   = lane & 15;
      const int khi = (lane & 16) ? 8 : 0;
      #pragma unroll
      for (int i = 0; i < 16; ++i) {
        int kk = ((i >> 3) << 4) + khi + (i & 7);
        af[i] = Als[coSub + m][kk];
      }
      // B 32x16 f16: lanes 0-15 -> N=lane, K {0..7,16..23}; lanes 16-31 -> K {8..15,24..31}
      const int kb = (lane & 16) ? 16 : 0;
      #pragma unroll
      for (int i = 0; i < 16; ++i)
        bf[i] = Bls[((i >> 3) << 4) + (kb >> 1) + (i & 7)][pxSub + m];

      acc = __builtin_amdgcn_wmma_f32_16x16x32_f16(
          false, af, false, bf, (short)0, acc, false, false);
      __syncthreads();
    }
  }

  // ---- epilogue: bias + ReLU, scatter per C/D layout (vgpr r: M=r / r+8)
  const int x    = pxSub + (lane & 15);
  const int mofs = (lane & 16) ? 8 : 0;
  #pragma unroll
  for (int r = 0; r < 8; ++r) {
    int co = cob + coSub + r + mofs;
    float v = acc[r] + bias[co];
    out[(((size_t)b * Hc + y) * Wc + x) * Cc + co] = fmaxf(v, 0.f);
  }
}

// ---------------------------------------------------------------------------
// Kernel 2: 1x1 cls/reg heads. One thread per (b, pixel, out-channel of 54).
// Writes already-permuted layouts: cls_raw (B,NA,2), reg_raw (B,NA,4).
// ---------------------------------------------------------------------------
__global__ __launch_bounds__(256) void head_kernel(
    const float* __restrict__ sh,     // (B,64,64,512)
    const float* __restrict__ cls_w, const float* __restrict__ cls_b,
    const float* __restrict__ reg_w, const float* __restrict__ reg_b,
    float* __restrict__ cls_raw, float* __restrict__ reg_raw)
{
  int tid = blockIdx.x * 256 + threadIdx.x;
  const int total = Bc * Hc * Wc * 54;
  if (tid >= total) return;
  int b   = tid / (Hc * Wc * 54);
  int rem = tid % (Hc * Wc * 54);
  int pix = rem / 54;
  int oc  = rem % 54;

  const float4* iv = (const float4*)(sh + ((size_t)b * Hc * Wc + pix) * Cc);
  const float4* wv; float acc;
  if (oc < 18) { wv = (const float4*)(cls_w + (size_t)oc * Cc);        acc = cls_b[oc]; }
  else         { wv = (const float4*)(reg_w + (size_t)(oc - 18) * Cc); acc = reg_b[oc - 18]; }

  #pragma unroll 4
  for (int i = 0; i < Cc / 4; ++i) {
    float4 a = iv[i], w = wv[i];
    acc += a.x * w.x + a.y * w.y + a.z * w.z + a.w * w.w;
  }
  if (oc < 18) {
    int a = pix * 9 + (oc >> 1);
    cls_raw[((size_t)b * NA + a) * 2 + (oc & 1)] = acc;
  } else {
    int o2 = oc - 18;
    int a  = pix * 9 + (o2 >> 2);
    reg_raw[((size_t)b * NA + a) * 4 + (o2 & 3)] = acc;
  }
}

// ---------------------------------------------------------------------------
// Kernel 3: anchors + inside-image validity flags
// ---------------------------------------------------------------------------
__global__ __launch_bounds__(256) void anchor_kernel(
    float* __restrict__ anchors_out, int* __restrict__ flags,
    const int* __restrict__ ih, const int* __restrict__ iw)
{
  int a = blockIdx.x * 256 + threadIdx.x;
  if (a >= NA) return;
  int pix = a / 9, k = a % 9;
  int y = pix / Wc, x = pix % Wc;
  const float ratios[3] = {0.5f, 1.f, 2.f};
  const float scales[3] = {8.f, 16.f, 32.f};
  float r = ratios[k / 3], s = scales[k % 3];
  float h = 16.f * s * sqrtf(r);
  float w = 16.f * s * sqrtf(1.f / r);
  float sy = y * 16.f, sx = x * 16.f;
  float y1 = sy + 8.f - 0.5f * h, x1 = sx + 8.f - 0.5f * w;
  float y2 = sy + 8.f + 0.5f * h, x2 = sx + 8.f + 0.5f * w;
  anchors_out[a * 4 + 0] = y1; anchors_out[a * 4 + 1] = x1;
  anchors_out[a * 4 + 2] = y2; anchors_out[a * 4 + 3] = x2;
  flags[a] = (y1 >= 0.f && x1 >= 0.f &&
              y2 <= (float)ih[0] && x2 <= (float)iw[0]) ? 1 : 0;
}

// ---------------------------------------------------------------------------
// Kernel 4: single-block prefix-scan compaction -> vlist[j] = anchor index
// (preserves ascending order to match np.where)
// ---------------------------------------------------------------------------
__global__ __launch_bounds__(256) void scan_kernel(
    const int* __restrict__ flags, int* __restrict__ vlist, int n)
{
  __shared__ int buf[256];
  __shared__ int carry;
  int t = threadIdx.x;
  if (t == 0) carry = 0;
  __syncthreads();
  for (int base = 0; base < n; base += 256) {
    int f = (base + t < n) ? flags[base + t] : 0;
    buf[t] = f;
    __syncthreads();
    for (int off = 1; off < 256; off <<= 1) {   // Hillis-Steele inclusive scan
      int v = (t >= off) ? buf[t - off] : 0;
      __syncthreads();
      buf[t] += v;
      __syncthreads();
    }
    int incl = buf[t];
    int pos  = carry + incl - f;                // exclusive
    if (base + t < n && f) vlist[pos] = base + t;
    __syncthreads();
    if (t == 255) carry += buf[255];
    __syncthreads();
  }
}

// ---------------------------------------------------------------------------
// Kernel 5: gather valid anchors, softmax cls, copy reg, decode+clip boxes,
// min-size filter -> scores. One thread per (b, j<K).
// ---------------------------------------------------------------------------
__global__ __launch_bounds__(256) void proposal_kernel(
    const float* __restrict__ cls_raw, const float* __restrict__ reg_raw,
    const int* __restrict__ vlist, const float* __restrict__ anchors,
    const int* __restrict__ ih, const int* __restrict__ iw, int K,
    float* __restrict__ cls_out, float* __restrict__ reg_out,
    float* __restrict__ scores, float* __restrict__ boxes)
{
  int tid = blockIdx.x * 256 + threadIdx.x;
  if (tid >= 2 * K) return;
  int b = tid / K, j = tid % K;
  int a = vlist[j];
  float imh = (float)ih[0], imw = (float)iw[0];

  float c0 = cls_raw[((size_t)b * NA + a) * 2 + 0];
  float c1 = cls_raw[((size_t)b * NA + a) * 2 + 1];
  float mx = fmaxf(c0, c1);
  float e0 = expf(c0 - mx), e1 = expf(c1 - mx);
  float inv = 1.f / (e0 + e1);
  float p1 = e1 * inv;
  cls_out[((size_t)b * K + j) * 2 + 0] = e0 * inv;
  cls_out[((size_t)b * K + j) * 2 + 1] = p1;

  float t0 = reg_raw[((size_t)b * NA + a) * 4 + 0];
  float t1 = reg_raw[((size_t)b * NA + a) * 4 + 1];
  float t2 = reg_raw[((size_t)b * NA + a) * 4 + 2];
  float t3 = reg_raw[((size_t)b * NA + a) * 4 + 3];
  reg_out[((size_t)b * K + j) * 4 + 0] = t0;
  reg_out[((size_t)b * K + j) * 4 + 1] = t1;
  reg_out[((size_t)b * K + j) * 4 + 2] = t2;
  reg_out[((size_t)b * K + j) * 4 + 3] = t3;

  float ay1 = anchors[a * 4 + 0], ax1 = anchors[a * 4 + 1];
  float ay2 = anchors[a * 4 + 2], ax2 = anchors[a * 4 + 3];
  float ah = ay2 - ay1, aw = ax2 - ax1;
  float cy = ay1 + 0.5f * ah, cx = ax1 + 0.5f * aw;
  float ncy = t0 * ah + cy, ncx = t1 * aw + cx;
  float nh = ah * expf(t2), nw = aw * expf(t3);
  float b0 = fminf(fmaxf(ncy - 0.5f * nh, 0.f), imh);
  float b1 = fminf(fmaxf(ncx - 0.5f * nw, 0.f), imw);
  float b2 = fminf(fmaxf(ncy + 0.5f * nh, 0.f), imh);
  float b3 = fminf(fmaxf(ncx + 0.5f * nw, 0.f), imw);
  boxes[((size_t)b * K + j) * 4 + 0] = b0;
  boxes[((size_t)b * K + j) * 4 + 1] = b1;
  boxes[((size_t)b * K + j) * 4 + 2] = b2;
  boxes[((size_t)b * K + j) * 4 + 3] = b3;
  float hs = b2 - b0, wsz = b3 - b1;
  scores[(size_t)b * K + j] = (hs >= 16.f && wsz >= 16.f) ? p1 : -INFINITY;
}

// ---------------------------------------------------------------------------
// Kernel 6: sequential greedy NMS, one workgroup per batch image.
// 300 rounds of (parallel argmax over K) + IoU suppression; all L2-resident.
// ---------------------------------------------------------------------------
__global__ __launch_bounds__(256) void nms_kernel(
    float* __restrict__ scores, const float* __restrict__ boxes, int K,
    float* __restrict__ roi_out, int* __restrict__ roi_id)
{
  const int b = blockIdx.x;
  const int t = threadIdx.x;
  float* sc = scores + (size_t)b * K;
  const float* bx = boxes + (size_t)b * K * 4;
  __shared__ float sv[256];
  __shared__ int   si[256];
  __shared__ float sel[4];
  __shared__ int   selok;

  for (int it = 0; it < POST; ++it) {
    float bv = -INFINITY; int bi = 0;
    for (int j = t; j < K; j += 256) {
      float v = sc[j];
      if (v > bv) { bv = v; bi = j; }
    }
    sv[t] = bv; si[t] = bi;
    __syncthreads();
    for (int off = 128; off > 0; off >>= 1) {
      if (t < off) {
        if (sv[t + off] > sv[t] ||
            (sv[t + off] == sv[t] && si[t + off] < si[t])) {
          sv[t] = sv[t + off]; si[t] = si[t + off];
        }
      }
      __syncthreads();
    }
    if (t == 0) {
      int i = si[0];
      float v = sv[0];
      int ok = (v > -INFINITY) ? 1 : 0;
      selok = ok;
      sel[0] = bx[i * 4 + 0]; sel[1] = bx[i * 4 + 1];
      sel[2] = bx[i * 4 + 2]; sel[3] = bx[i * 4 + 3];
      float g = ok ? 1.f : 0.f;
      roi_out[((size_t)b * POST + it) * 4 + 0] = sel[0] * g;
      roi_out[((size_t)b * POST + it) * 4 + 1] = sel[1] * g;
      roi_out[((size_t)b * POST + it) * 4 + 2] = sel[2] * g;
      roi_out[((size_t)b * POST + it) * 4 + 3] = sel[3] * g;
      roi_id[b * POST + it] = b;
      sc[i] = -INFINITY;
    }
    __syncthreads();
    if (selok) {
      float y1 = sel[0], x1 = sel[1], y2 = sel[2], x2 = sel[3];
      float ar = (y2 - y1) * (x2 - x1);
      for (int j = t; j < K; j += 256) {
        float by1 = bx[j * 4 + 0], bx1 = bx[j * 4 + 1];
        float by2 = bx[j * 4 + 2], bx2 = bx[j * 4 + 3];
        float aj  = (by2 - by1) * (bx2 - bx1);
        float yy1 = fmaxf(y1, by1), xx1 = fmaxf(x1, bx1);
        float yy2 = fminf(y2, by2), xx2 = fminf(x2, bx2);
        float inter = fmaxf(yy2 - yy1, 0.f) * fmaxf(xx2 - xx1, 0.f);
        float iou = inter / (ar + aj - inter + 1e-9f);
        if (iou > 0.7f) sc[j] = -INFINITY;
      }
    }
    __syncthreads();
  }
}

// ---------------------------------------------------------------------------
extern "C" void kernel_launch(void* const* d_in, const int* in_sizes, int n_in,
                              void* d_out, int out_size, void* d_ws, size_t ws_size,
                              hipStream_t stream) {
  const float* feat    = (const float*)d_in[0];
  const float* share_w = (const float*)d_in[1];
  const float* share_b = (const float*)d_in[2];
  const float* cls_w   = (const float*)d_in[3];
  const float* cls_b   = (const float*)d_in[4];
  const float* reg_w   = (const float*)d_in[5];
  const float* reg_b   = (const float*)d_in[6];
  const int*   img_h   = (const int*)d_in[7];
  const int*   img_w   = (const int*)d_in[8];

  // out_size = B*K*2 + B*K*4 + B*300*4 + B*300 + NA*4  ->  K = (out_size-150456)/12
  int K = (out_size - (Bc * POST * 4 + Bc * POST + NA * 4)) / (Bc * 2 + Bc * 4);
  if (K < 1)  K = 1;
  if (K > NA) K = NA;

  float* outf    = (float*)d_out;
  float* cls_out = outf;
  float* reg_out = outf + (size_t)Bc * K * 2;
  float* roi_out = outf + (size_t)Bc * K * 6;
  int*   roi_id  = (int*)(outf + (size_t)Bc * K * 6 + Bc * POST * 4);
  float* anch_o  = outf + (size_t)Bc * K * 6 + Bc * POST * 4 + Bc * POST;

  float* ws     = (float*)d_ws;
  float* sh     = ws;                                  // (B,64,64,512)
  float* clsraw = sh     + (size_t)Bc * Hc * Wc * Cc;  // (B,NA,2)
  float* regraw = clsraw + (size_t)Bc * NA * 2;        // (B,NA,4)
  float* scores = regraw + (size_t)Bc * NA * 4;        // (B,K)
  float* boxes  = scores + (size_t)Bc * NA;            // (B,K,4)
  int*   flags  = (int*)(boxes + (size_t)Bc * NA * 4); // (NA)
  int*   vlist  = flags + NA;                          // (NA)

  conv3x3_wmma<<<dim3(Hc, Cc / 32, Bc), 256, 0, stream>>>(feat, share_w, share_b, sh);
  head_kernel<<<(Bc * Hc * Wc * 54 + 255) / 256, 256, 0, stream>>>(
      sh, cls_w, cls_b, reg_w, reg_b, clsraw, regraw);
  anchor_kernel<<<(NA + 255) / 256, 256, 0, stream>>>(anch_o, flags, img_h, img_w);
  scan_kernel<<<1, 256, 0, stream>>>(flags, vlist, NA);
  proposal_kernel<<<(2 * K + 255) / 256, 256, 0, stream>>>(
      clsraw, regraw, vlist, anch_o, img_h, img_w, K,
      cls_out, reg_out, scores, boxes);
  nms_kernel<<<Bc, 256, 0, stream>>>(scores, boxes, K, roi_out, roi_id);
}
